// ArcFace_59356448030740
// MI455X (gfx1250) — matrix operand order repaired and checked by reading, not compiled, not run on hardware
//
#include <hip/hip_runtime.h>
#include <hip/hip_bf16.h>
#include <math.h>

// ---------------- problem constants (from reference) ----------------
#define S_SCALE  64.0f
#define MARGIN   0.5f
#define EPS_C    1e-7f
#define NORM_EPS 1e-12f
#define B_DIM    512
#define D_DIM    512
#define C_DIM    64000

// ---------------- CDNA5 WMMA types ----------------
typedef __bf16 bf16_t;
typedef bf16_t v16bf __attribute__((ext_vector_type(16)));
typedef bf16_t v8bf  __attribute__((ext_vector_type(8)));
typedef float  v8f   __attribute__((ext_vector_type(8)));

// Wave tile: MT*16 batch rows x CT*16 class cols per wave
#define MT 2
#define CT 4
#define WAVES_PER_BLOCK 8   // block tile: 32 batch x 512 classes

// =====================================================================
// Kernel 1: row L2 norms + bf16 pre-conversion.  One wave32 per row,
// each lane owns 16 contiguous elements (32 lanes * 16 = 512 = D).
//   waves [0, B)     -> xn = x/max(||x||,eps)  (fp32) and xnbf = bf16(xn)
//   waves [B, B + C) -> winv[c] = 1/max(||W_c||,eps) and wbf = bf16(W_c)
// =====================================================================
__global__ void __launch_bounds__(256) arcface_norms(
    const float* __restrict__ x, const float* __restrict__ W,
    float* __restrict__ xn, float* __restrict__ winv,
    bf16_t* __restrict__ xnbf, bf16_t* __restrict__ wbf) {
  const int tid  = blockIdx.x * blockDim.x + threadIdx.x;
  const int wave = tid >> 5;
  const int lane = tid & 31;

  if (wave < B_DIM) {
    const float* p = x + (size_t)wave * D_DIM + lane * 16;
    float v[16];
    float ss = 0.0f;
#pragma unroll
    for (int i = 0; i < 16; ++i) { v[i] = p[i]; ss = fmaf(v[i], v[i], ss); }
#pragma unroll
    for (int m = 16; m >= 1; m >>= 1) ss += __shfl_xor(ss, m, 32);
    const float inv = 1.0f / fmaxf(sqrtf(ss), NORM_EPS);

    float* xo = xn + (size_t)wave * D_DIM + lane * 16;
    alignas(16) bf16_t tb[16];
#pragma unroll
    for (int i = 0; i < 16; ++i) {
      const float s = v[i] * inv;
      xo[i] = s;
      tb[i] = (bf16_t)s;
    }
    uint4* d = reinterpret_cast<uint4*>(xnbf + (size_t)wave * D_DIM + lane * 16);
    const uint4* s4 = reinterpret_cast<const uint4*>(tb);
    d[0] = s4[0]; d[1] = s4[1];
  } else if (wave < B_DIM + C_DIM) {
    const int c = wave - B_DIM;
    const float* p = W + (size_t)c * D_DIM + lane * 16;
    float v[16];
    float ss = 0.0f;
#pragma unroll
    for (int i = 0; i < 16; ++i) { v[i] = p[i]; ss = fmaf(v[i], v[i], ss); }
#pragma unroll
    for (int m = 16; m >= 1; m >>= 1) ss += __shfl_xor(ss, m, 32);
    if (lane == 0) winv[c] = 1.0f / fmaxf(sqrtf(ss), NORM_EPS);

    alignas(16) bf16_t tb[16];
#pragma unroll
    for (int i = 0; i < 16; ++i) tb[i] = (bf16_t)v[i];   // raw W in bf16
    uint4* d = reinterpret_cast<uint4*>(wbf + (size_t)c * D_DIM + lane * 16);
    const uint4* s4 = reinterpret_cast<const uint4*>(tb);
    d[0] = s4[0]; d[1] = s4[1];
  }
}

// =====================================================================
// Fragment loaders (bf16 16x16x32 layouts per CDNA5 ISA 7.12.2),
// loading pre-converted bf16 directly -> 2x global_load_b128 each.
// =====================================================================
// A: 16x32 (MxK). lane: M = lane&15, half = lane>>4.
//    e<8 : K = 8*half + e ; e>=8 : K = 8*half + 8 + e
__device__ __forceinline__ v16bf load_a_frag(const bf16_t* __restrict__ row, int half) {
  const bf16_t* p = row + 8 * half;
  const v8bf lo = *reinterpret_cast<const v8bf*>(p);
  const v8bf hi = *reinterpret_cast<const v8bf*>(p + 16);
  return __builtin_shufflevector(lo, hi, 0, 1, 2, 3, 4, 5, 6, 7,
                                 8, 9, 10, 11, 12, 13, 14, 15);
}
// B: 32x16 (KxN). lane: N = lane&15, half = lane>>4. e<8 : K = 8*half + e ; e>=8 : K = 8*half + 8 + e
// -> two runs of 8 bf16, 16 apart, of a (row-major) W row.
__device__ __forceinline__ v16bf load_b_frag(const bf16_t* __restrict__ p) {
  const v8bf lo = *reinterpret_cast<const v8bf*>(p);
  const v8bf hi = *reinterpret_cast<const v8bf*>(p + 16);
  return __builtin_shufflevector(lo, hi, 0, 1, 2, 3, 4, 5, 6, 7,
                                 8, 9, 10, 11, 12, 13, 14, 15);
}

// =====================================================================
// Kernel 2: bf16 WMMA GEMM + fused exp-sum reduction.
//   theta[b,c] = (xnbf_b . wbf_c) * winv[c]   (fp32 accumulate)
//   excl[b]   += sum_c exp(S * theta[b,c])    (all classes; target
//                column subtracted in finish kernel)
// Grid: x = C/512 class groups, y = B/32 batch groups. Block = 256.
// =====================================================================
__global__ void __launch_bounds__(256) arcface_gemm(
    const bf16_t* __restrict__ xnbf, const bf16_t* __restrict__ wbf,
    const float* __restrict__ winv, float* __restrict__ excl) {
  const int lane = threadIdx.x & 31;
  const int wid  = threadIdx.x >> 5;
  const int half = lane >> 4;
  const int nl   = lane & 15;

  const int b0 = blockIdx.y * (MT * 16);
  const int c0 = blockIdx.x * (WAVES_PER_BLOCK * CT * 16) + wid * (CT * 16);

  const bf16_t* arow[MT];
#pragma unroll
  for (int mt = 0; mt < MT; ++mt)
    arow[mt] = xnbf + (size_t)(b0 + mt * 16 + nl) * D_DIM;
  const bf16_t* brow[CT];
#pragma unroll
  for (int ct = 0; ct < CT; ++ct)
    brow[ct] = wbf + (size_t)(c0 + ct * 16 + nl) * D_DIM + 8 * half;

  v8f acc[MT][CT];
#pragma unroll
  for (int mt = 0; mt < MT; ++mt)
#pragma unroll
    for (int ct = 0; ct < CT; ++ct) acc[mt][ct] = {};

#pragma unroll 2
  for (int kc = 0; kc < D_DIM; kc += 32) {
    v16bf a[MT];
#pragma unroll
    for (int mt = 0; mt < MT; ++mt) a[mt] = load_a_frag(arow[mt] + kc, half);
#pragma unroll
    for (int ct = 0; ct < CT; ++ct) {
      const v16bf bfr = load_b_frag(brow[ct] + kc);
#pragma unroll
      for (int mt = 0; mt < MT; ++mt) {
        acc[mt][ct] = __builtin_amdgcn_wmma_f32_16x16x32_bf16(
            /*neg_a=*/false, a[mt], /*neg_b=*/false, bfr,
            /*c_mod=*/(short)0, acc[mt][ct], /*reuse_a=*/false, /*reuse_b=*/false);
      }
    }
  }

  // ---- epilogue: theta -> exp(S*theta) -> row sums -> atomic add ----
  float iw[CT];
#pragma unroll
  for (int ct = 0; ct < CT; ++ct) iw[ct] = winv[c0 + ct * 16 + nl];

#pragma unroll
  for (int mt = 0; mt < MT; ++mt) {
    float rowsum[8];
#pragma unroll
    for (int r = 0; r < 8; ++r) rowsum[r] = 0.0f;
#pragma unroll
    for (int ct = 0; ct < CT; ++ct)
#pragma unroll
      for (int r = 0; r < 8; ++r)
        rowsum[r] += __expf(S_SCALE * acc[mt][ct][r] * iw[ct]);

    // D layout: VGPR r holds row m = r + 8*half, col n = lane&15.
#pragma unroll
    for (int r = 0; r < 8; ++r) {
      float v = rowsum[r];
      v += __shfl_xor(v, 1, 32);
      v += __shfl_xor(v, 2, 32);
      v += __shfl_xor(v, 4, 32);
      v += __shfl_xor(v, 8, 32);
      if (nl == 0) atomicAdd(&excl[b0 + mt * 16 + 8 * half + r], v);
    }
  }
}

// =====================================================================
// Kernel 3: per-sample finish + mean.  One thread per batch row, 1 block.
//   - exact fp32 target logit (numerator is not error-averaged -> fp32)
//   - bf16-matched target exp subtracted from accumulated class sum
//     ((bf16)xn[k] and (bf16)W[k] are bit-identical to xnbf/wbf)
// =====================================================================
__global__ void __launch_bounds__(512) arcface_finish(
    const float* __restrict__ xn, const float* __restrict__ W,
    const float* __restrict__ winv, const float* __restrict__ excl,
    const int* __restrict__ y, float* __restrict__ out) {
  const int b  = threadIdx.x;
  const int yb = y[b];
  const float* xr = xn + (size_t)b * D_DIM;
  const float* wr = W + (size_t)yb * D_DIM;

  float dot = 0.0f, dotbf = 0.0f;
#pragma unroll 4
  for (int k = 0; k < D_DIM; ++k) {
    dot = fmaf(xr[k], wr[k], dot);
    dotbf += (float)(bf16_t)xr[k] * (float)(bf16_t)wr[k];
  }
  const float iw      = winv[yb];
  const float t_exact = dot * iw;
  const float t_bf    = dotbf * iw;

  // exclude target column (bf16-matched, same __expf as GEMM epilogue)
  const float e_sum = excl[b] - __expf(S_SCALE * t_bf);

  const float tc  = fminf(fmaxf(t_exact, -1.0f + EPS_C), 1.0f - EPS_C);
  const float num = S_SCALE * cosf(acosf(tc) + MARGIN);
  const float den = expf(num) + e_sum;
  float v = logf(den) - num;          // = -(num - log(den))

  // block reduce (16 waves of 32)
  const int lane = threadIdx.x & 31;
  const int wid  = threadIdx.x >> 5;
#pragma unroll
  for (int m = 16; m >= 1; m >>= 1) v += __shfl_xor(v, m, 32);
  __shared__ float red[16];
  if (lane == 0) red[wid] = v;
  __syncthreads();
  if (threadIdx.x == 0) {
    float s = 0.0f;
#pragma unroll
    for (int i = 0; i < 16; ++i) s += red[i];
    out[0] = s * (1.0f / (float)B_DIM);
  }
}

// =====================================================================
extern "C" void kernel_launch(void* const* d_in, const int* in_sizes, int n_in,
                              void* d_out, int out_size, void* d_ws, size_t ws_size,
                              hipStream_t stream) {
  (void)in_sizes; (void)n_in; (void)out_size; (void)ws_size;
  const float* x = (const float*)d_in[0];
  const int*   y = (const int*)d_in[1];
  const float* W = (const float*)d_in[2];

  char* ws = (char*)d_ws;
  float*  xn   = (float*)ws;                               // B*D fp32 (1 MB)
  float*  winv = (float*)(ws + (size_t)1048576);           // C fp32 (256 KB)
  float*  excl = (float*)(ws + (size_t)1048576 + 256000);  // B fp32 (2 KB)
  bf16_t* xnbf = (bf16_t*)(ws + (size_t)1048576 + 256000 + 2048);            // B*D bf16
  bf16_t* wbf  = (bf16_t*)(ws + (size_t)1048576 + 256000 + 2048 + 524288);   // C*D bf16 (64 MB)

  hipMemsetAsync(excl, 0, B_DIM * sizeof(float), stream);

  {
    const int total_threads = (B_DIM + C_DIM) * 32;
    const int threads = 256;
    const int blocks  = (total_threads + threads - 1) / threads;
    arcface_norms<<<blocks, threads, 0, stream>>>(x, W, xn, winv, xnbf, wbf);
  }
  {
    dim3 grid(C_DIM / (WAVES_PER_BLOCK * CT * 16),  // 125
              B_DIM / (MT * 16));                   // 16
    arcface_gemm<<<grid, 256, 0, stream>>>(xnbf, wbf, winv, excl);
  }
  arcface_finish<<<1, 512, 0, stream>>>(xn, W, winv, excl, y, (float*)d_out);
}
